// TemporalEigenstateNetwork_154618823407
// MI455X (gfx1250) — matrix operand order repaired and checked by reading, not compiled, not run on hardware
//
#include <hip/hip_runtime.h>
#include <hip/hip_bf16.h>

typedef __attribute__((ext_vector_type(16))) __bf16 v16bf;
typedef __attribute__((ext_vector_type(8)))  float  v8f;
typedef __attribute__((ext_vector_type(8)))  unsigned short u16x8;
typedef __attribute__((ext_vector_type(16))) unsigned short u16x16;

#define V_  50257
#define D_  1024
#define L_  6
#define C_  2
#define K_  128
#define T_  2048
#define FF_ 2048
#define B_  2
#define M_  (B_*T_)   /* 4096 rows of activations */

// ---------------------------------------------------------------- helpers

__device__ __forceinline__ unsigned short f2bf(float f){
  unsigned u = __builtin_bit_cast(unsigned, f);
  u += 0x7FFFu + ((u >> 16) & 1u);          // round-to-nearest-even
  return (unsigned short)(u >> 16);
}

__device__ __forceinline__ v8f wmma_bf16(u16x16 a, u16x16 b, v8f c){
  return __builtin_amdgcn_wmma_f32_16x16x32_bf16(
      false, __builtin_bit_cast(v16bf, a),
      false, __builtin_bit_cast(v16bf, b),
      (short)0, c, false, false);
}

// A-matrix (16x32, M x K) per-lane layout: lane l holds row m=l&15,
// K = kb..kb+7 (elems 0-7) and 16+kb..16+kb+7 (elems 8-15), kb=(l>>4)*8.
__device__ __forceinline__ u16x16 load_fragA(const unsigned short* row, int kb){
  u16x8 lo = *(const u16x8*)(row + kb);
  u16x8 hi = *(const u16x8*)(row + kb + 16);
  return __builtin_shufflevector(lo, hi, 0,1,2,3,4,5,6,7,8,9,10,11,12,13,14,15);
}

// CDNA5 async copy: global -> LDS, 16B per lane, tracked by ASYNCcnt.
__device__ __forceinline__ void async_g2l_b128(unsigned lds_off, const void* gaddr){
  asm volatile("global_load_async_to_lds_b128 %0, %1, off"
               :: "v"(lds_off), "v"(gaddr) : "memory");
}
__device__ __forceinline__ void wait_asynccnt0(){
  asm volatile("s_wait_asynccnt 0x0" ::: "memory");
}

// ---------------------------------------------------------------- conversions

__global__ __launch_bounds__(256)
void f32_to_bf16_kernel(const float* __restrict__ in, unsigned short* __restrict__ out, long n){
  long i = (long)blockIdx.x * blockDim.x + threadIdx.x;
  long stride = (long)gridDim.x * blockDim.x;
  for (; i < n; i += stride) out[i] = f2bf(in[i]);
}

// ---------------------------------------------------------------- embedding

__global__ __launch_bounds__(256)
void embed_kernel(const int* __restrict__ tokens, const float* __restrict__ temb,
                  const float* __restrict__ pemb, float* __restrict__ x,
                  unsigned short* __restrict__ xb){
  int r = blockIdx.x;                // r = b*T + t
  int t = r & (T_ - 1);
  int tok = tokens[r];
  const float* te = temb + (long)tok * D_;
  const float* pe = pemb + (long)t * D_;
  long off = (long)r * D_;
  for (int d = threadIdx.x; d < D_; d += 256){
    float v = te[d] + pe[d];
    x[off + d] = v;
    xb[off + d] = f2bf(v);
  }
}

// ---------------------------------------------------------------- WMMA GEMM
// C[M,N](f32) = A[M,Kd](bf16 rows) * B[N,Kd](bf16 rows)^T
// Block = 8 waves sharing one 64-wide N strip; the 64x32 B chunk per k-step is
// staged into LDS once per block with GLOBAL_LOAD_ASYNC_TO_LDS_B128 (double
// buffered, issued one step ahead so the DMA hides under the WMMA chain).
// amap: 0 identity; 1: out row m=(t,b) interleaved -> A row b*T+t  (beta)
//       2: out row m=b*T+t -> A row t*B+b                          (out-proj)

#define BROW 40   /* padded row stride (u16) -> bank-conflict-free frag reads */

__global__ __launch_bounds__(256)
void gemm_bf16_wmma(const unsigned short* __restrict__ A, int lda, long sAz,
                    const unsigned short* __restrict__ Bw, int ldb, long sBz,
                    float* __restrict__ Co, int ldc, long sCz,
                    int M, int N, int Kd, int amap)
{
  const int lane = threadIdx.x & 31;
  const int wv   = threadIdx.x >> 5;
  const int tid  = threadIdx.x;
  A  += blockIdx.z * sAz;
  Bw += blockIdx.z * sBz;
  Co += blockIdx.z * sCz;
  const int m0 = (blockIdx.y * 8 + wv) * 32;
  const int n0 = blockIdx.x * 64;
  const int lm  = lane & 15;
  const int kb  = (lane >> 4) * 8;
  const int k16 = (lane >> 4) * 16;

  __shared__ unsigned short sB[2][64][BROW];

  // this thread's slice of the B tile: row 0..63, 16B segment 0..3
  const int srow = tid >> 2;
  const int sseg = tid & 3;
  int brow = n0 + srow; if (brow >= N) brow = N - 1;           // N-edge clamp
  const unsigned short* gB = Bw + (long)brow * ldb + sseg * 8; // + k0 later
  unsigned ldsB0 = (unsigned)(unsigned long long)&sB[0][srow][sseg * 8];
  unsigned ldsB1 = (unsigned)(unsigned long long)&sB[1][srow][sseg * 8];

  int r0 = m0 + lm, r1 = m0 + 16 + lm;
  if (amap == 1){
    int t = r0 >> 1, b = r0 & 1; r0 = b * T_ + t;
    t = r1 >> 1; b = r1 & 1;     r1 = b * T_ + t;
  } else if (amap == 2){
    int b = r0 >> 11, t = r0 & (T_ - 1); r0 = t * B_ + b;
    b = r1 >> 11; t = r1 & (T_ - 1);     r1 = t * B_ + b;
  }
  const unsigned short* a0 = A + (long)r0 * lda;
  const unsigned short* a1 = A + (long)r1 * lda;

  v8f acc[2][4];
#pragma unroll
  for (int i = 0; i < 2; ++i)
#pragma unroll
    for (int j = 0; j < 4; ++j) acc[i][j] = (v8f){0,0,0,0,0,0,0,0};

  // prologue: stage k0 = 0 into buffer 0
  async_g2l_b128(ldsB0, gB);
  wait_asynccnt0();
  __syncthreads();

  int buf = 0;
  for (int k0 = 0; k0 < Kd; k0 += 32){
    // issue next-chunk DMA before computing (overlaps with WMMA below)
    if (k0 + 32 < Kd)
      async_g2l_b128(buf ? ldsB0 : ldsB1, gB + k0 + 32);

    __builtin_prefetch(a0 + k0 + 128, 0, 1);
    u16x16 af0 = load_fragA(a0 + k0, kb);
    u16x16 af1 = load_fragA(a1 + k0, kb);
#pragma unroll
    for (int j = 0; j < 4; ++j){
      const unsigned short* br = &sB[buf][j * 16 + lm][kb];
      u16x8 lo = *(const u16x8*)br;
      u16x8 hi = *(const u16x8*)(br + 16);
      u16x16 bf = __builtin_shufflevector(lo, hi,
          0,1,2,3,4,5,6,7,8,9,10,11,12,13,14,15);
      acc[0][j] = wmma_bf16(af0, bf, acc[0][j]);
      acc[1][j] = wmma_bf16(af1, bf, acc[1][j]);
    }
    wait_asynccnt0();      // next-chunk DMA done (only op outstanding)
    __syncthreads();       // all waves done reading buf; writes visible
    buf ^= 1;
  }

  if (m0 >= M) return;     // (kept after barriers: wave-uniform, M always %32)

  // D layout: elem e -> row m0+i*16+e (lanes 0-15) or +8 (lanes 16-31); col n0+j*16+lm
  const int rb8 = (lane >> 4) * 8;
#pragma unroll
  for (int i = 0; i < 2; ++i){
#pragma unroll
    for (int j = 0; j < 4; ++j){
      int cc = n0 + j * 16 + lm;
      if (cc < N){
        long base = (long)(m0 + i * 16 + rb8) * ldc + cc;
#pragma unroll
        for (int e = 0; e < 8; ++e) Co[base + (long)e * ldc] = acc[i][j][e];
      }
    }
  }
}

// ---------------------------------------------------------------- eigenstate scan
// One block per cell, 8 waves. Wave w owns output columns [16w,16w+16).
// State (sr,si)[B=2][128]; per step: decay/rotate + beta, then matmul by R
// done as one 16x128 (rows 0-3 live) x 128x128 WMMA chain (4 k-steps/wave).

__global__ __launch_bounds__(256)
void scan_kernel(const float* __restrict__ alpha, const float* __restrict__ omega,
                 const unsigned short* __restrict__ rb, const float* __restrict__ beta,
                 unsigned short* __restrict__ ysb)
{
  const int c = blockIdx.x;
  alpha += c * K_; omega += c * K_;
  rb   += (long)c * K_ * K_;
  beta += (long)c * T_ * B_ * K_;
  ysb  += (long)c * T_ * B_ * K_;

  const int lane = threadIdx.x & 31;
  const int wv   = threadIdx.x >> 5;
  const int lm   = lane & 15;
  const int kb   = (lane >> 4) * 8;
  const int k16  = (lane >> 4) * 16;
  const int col  = wv * 16 + lm;          // this lane's output column

  __shared__ unsigned short sA[16][K_];   // padded 16x128 A tile (rows 4-15 stay 0)
  for (int i = threadIdx.x; i < 16 * K_; i += 256) ((unsigned short*)sA)[i] = 0;

  // per-lane recurrence constants
  const float mag = 1.f / (1.f + __expf(-alpha[col]));
  const float cp = __cosf(omega[col]);
  const float sp = __sinf(omega[col]);

  // preload this wave's 4 resident B fragments of R (K x N strip)
  u16x16 bfrag[4];
#pragma unroll
  for (int kk = 0; kk < 4; ++kk){
    u16x16 f;
#pragma unroll
    for (int e = 0; e < 16; ++e){
      int Kidx = kk * 32 + kb + (e & 7) + (e >> 3) * 16;
      f[e] = rb[(long)Kidx * K_ + col];
    }
    bfrag[kk] = f;
  }
  __syncthreads();

  float sr0 = 0.f, sr1 = 0.f, si0 = 0.f, si1 = 0.f;

  for (int t = 0; t < T_; ++t){
    const float* bt = beta + (long)t * (B_ * K_);
    float b0 = bt[col], b1 = bt[K_ + col];
    float nr0 = mag * (sr0 * cp - si0 * sp) + b0;
    float nr1 = mag * (sr1 * cp - si1 * sp) + b1;
    float ni0 = mag * (sr0 * sp + si0 * cp);
    float ni1 = mag * (sr1 * sp + si1 * cp);
    if (lane < 16){
      sA[0][col] = f2bf(nr0);
      sA[1][col] = f2bf(nr1);
      sA[2][col] = f2bf(ni0);
      sA[3][col] = f2bf(ni1);
    }
    __syncthreads();

    v8f acc = (v8f){0,0,0,0,0,0,0,0};
#pragma unroll
    for (int kk = 0; kk < 4; ++kk){
      u16x8 lo = *(const u16x8*)&sA[lm][kk * 32 + kb];
      u16x8 hi = *(const u16x8*)&sA[lm][kk * 32 + kb + 16];
      u16x16 af = __builtin_shufflevector(lo, hi,
          0,1,2,3,4,5,6,7,8,9,10,11,12,13,14,15);
      acc = wmma_bf16(af, bfrag[kk], acc);
    }
    __syncthreads();               // all A-frag LDS reads done before next stores

    sr0 = acc[0]; sr1 = acc[1]; si0 = acc[2]; si1 = acc[3];  // rows 0-3 (lanes 0-15)
    if (lane < 16){
      ysb[(long)(t * 2 + 0) * K_ + col] = f2bf(sr0);
      ysb[(long)(t * 2 + 1) * K_ + col] = f2bf(sr1);
    }
  }
}

// ---------------------------------------------------------------- layernorm (fused)
// y = LN(base + addScale*add) * w + b ; outF (=|+=) y ; optional bf16 mirror.

__global__ __launch_bounds__(256)
void ln_fused(const float* __restrict__ base, const float* __restrict__ add, float addScale,
              const float* __restrict__ w, const float* __restrict__ b,
              float* __restrict__ outF, unsigned short* __restrict__ outB, int accumulate)
{
  const int r = blockIdx.x;
  const long off = (long)r * D_;
  __shared__ float s1[256], s2[256];
  float v[4]; float s = 0.f, ss = 0.f;
#pragma unroll
  for (int i = 0; i < 4; ++i){
    int d = threadIdx.x + i * 256;
    float t = base[off + d];
    if (add) t += addScale * add[off + d];
    v[i] = t; s += t; ss += t * t;
  }
  s1[threadIdx.x] = s; s2[threadIdx.x] = ss;
  __syncthreads();
  for (int st = 128; st > 0; st >>= 1){
    if (threadIdx.x < st){
      s1[threadIdx.x] += s1[threadIdx.x + st];
      s2[threadIdx.x] += s2[threadIdx.x + st];
    }
    __syncthreads();
  }
  const float mean = s1[0] * (1.f / D_);
  const float var  = s2[0] * (1.f / D_) - mean * mean;
  const float inv  = rsqrtf(var + 1e-5f);
#pragma unroll
  for (int i = 0; i < 4; ++i){
    int d = threadIdx.x + i * 256;
    float y = (v[i] - mean) * inv * w[d] + b[d];
    if (accumulate) outF[off + d] += y; else outF[off + d] = y;
    if (outB) outB[off + d] = f2bf(y);
  }
}

// ---------------------------------------------------------------- swiglu

__global__ __launch_bounds__(256)
void swiglu_kernel(const float* __restrict__ g, const float* __restrict__ u,
                   unsigned short* __restrict__ hb, long n){
  long i = (long)blockIdx.x * blockDim.x + threadIdx.x;
  long stride = (long)gridDim.x * blockDim.x;
  for (; i < n; i += stride){
    float gv = g[i];
    float sv = gv / (1.f + __expf(-gv));
    hb[i] = f2bf(sv * u[i]);
  }
}

// ---------------------------------------------------------------- host

extern "C" void kernel_launch(void* const* d_in, const int* in_sizes, int n_in,
                              void* d_out, int out_size, void* d_ws, size_t ws_size,
                              hipStream_t stream)
{
  (void)in_sizes; (void)n_in; (void)out_size; (void)ws_size;
  const int*   tokens    = (const int*)  d_in[0];
  const float* token_emb = (const float*)d_in[1];
  const float* pos_emb   = (const float*)d_in[2];
  const float* alpha     = (const float*)d_in[3];
  const float* omega     = (const float*)d_in[4];
  const float* in_proj   = (const float*)d_in[5];
  const float* out_proj  = (const float*)d_in[6];
  const float* resonance = (const float*)d_in[7];
  const float* cln_w     = (const float*)d_in[8];
  const float* cln_b     = (const float*)d_in[9];
  const float* ffn_gate  = (const float*)d_in[10];
  const float* ffn_up    = (const float*)d_in[11];
  const float* ffn_down  = (const float*)d_in[12];
  const float* ln1_w = (const float*)d_in[13];
  const float* ln1_b = (const float*)d_in[14];
  const float* ln2_w = (const float*)d_in[15];
  const float* ln2_b = (const float*)d_in[16];
  const float* fin_w = (const float*)d_in[17];
  const float* fin_b = (const float*)d_in[18];
  float* logits = (float*)d_out;

  char* ws = (char*)d_ws;
  size_t off = 0;
  auto alloc = [&](size_t bytes)->char*{
    char* p = ws + off; off = (off + bytes + 255) & ~(size_t)255; return p;
  };
  unsigned short* temb_b = (unsigned short*)alloc((size_t)V_ * D_ * 2);
  unsigned short* inp_b  = (unsigned short*)alloc((size_t)L_ * C_ * K_ * D_ * 2);
  unsigned short* outp_b = (unsigned short*)alloc((size_t)L_ * C_ * D_ * K_ * 2);
  unsigned short* rb     = (unsigned short*)alloc((size_t)L_ * C_ * K_ * K_ * 2);
  unsigned short* fg_b   = (unsigned short*)alloc((size_t)L_ * FF_ * D_ * 2);
  unsigned short* fu_b   = (unsigned short*)alloc((size_t)L_ * FF_ * D_ * 2);
  unsigned short* fd_b   = (unsigned short*)alloc((size_t)L_ * D_ * FF_ * 2);
  float*          x      = (float*)alloc((size_t)M_ * D_ * 4);
  unsigned short* xb     = (unsigned short*)alloc((size_t)M_ * D_ * 2);
  float*          mixed  = (float*)alloc((size_t)M_ * D_ * 4);
  float*          betaw  = (float*)alloc((size_t)C_ * T_ * B_ * K_ * 4);
  unsigned short* ysb    = (unsigned short*)alloc((size_t)C_ * T_ * B_ * K_ * 2);
  float*          ctmp   = (float*)alloc((size_t)C_ * M_ * D_ * 4);  // also reused as FFN 'y'
  float*          gbuf   = (float*)alloc((size_t)M_ * FF_ * 4);
  float*          ubuf   = (float*)alloc((size_t)M_ * FF_ * 4);
  unsigned short* hb     = (unsigned short*)alloc((size_t)M_ * FF_ * 2);

  // ---- weight conversion to bf16 (deterministic, every call)
  auto conv = [&](const float* src, unsigned short* dst, long n){
    f32_to_bf16_kernel<<<2048, 256, 0, stream>>>(src, dst, n);
  };
  conv(token_emb, temb_b, (long)V_ * D_);
  conv(in_proj,   inp_b,  (long)L_ * C_ * K_ * D_);
  conv(out_proj,  outp_b, (long)L_ * C_ * D_ * K_);
  conv(resonance, rb,     (long)L_ * C_ * K_ * K_);
  conv(ffn_gate,  fg_b,   (long)L_ * FF_ * D_);
  conv(ffn_up,    fu_b,   (long)L_ * FF_ * D_);
  conv(ffn_down,  fd_b,   (long)L_ * D_ * FF_);

  // ---- embedding
  embed_kernel<<<M_, 256, 0, stream>>>(tokens, token_emb, pos_emb, x, xb);

  const dim3 blk(256);
  for (int l = 0; l < L_; ++l){
    // beta[c,t,b,k] = x @ Win[l,c]^T   (A rows remapped bt->tb)
    gemm_bf16_wmma<<<dim3(K_ / 64, M_ / 256, C_), blk, 0, stream>>>(
        xb, D_, 0,
        inp_b + (size_t)l * C_ * K_ * D_, D_, (long)K_ * D_,
        betaw, K_, (long)T_ * B_ * K_,
        M_, K_, D_, 1);

    // sequential eigenstate scan (one block per cell), emits ys as bf16
    scan_kernel<<<dim3(C_), blk, 0, stream>>>(
        alpha + (size_t)l * C_ * K_, omega + (size_t)l * C_ * K_,
        rb + (size_t)l * C_ * K_ * K_, betaw, ysb);

    // out[c] = ys @ Wout[l,c]^T  (A rows remapped tb->bt)
    gemm_bf16_wmma<<<dim3(D_ / 64, M_ / 256, C_), blk, 0, stream>>>(
        ysb, K_, (long)T_ * B_ * K_,
        outp_b + (size_t)l * C_ * D_ * K_, K_, (long)D_ * K_,
        ctmp, D_, (long)M_ * D_,
        M_, D_, K_, 2);

    // mixed = LN_c0(out0); mixed += LN_c1(out1)
    for (int c = 0; c < C_; ++c){
      ln_fused<<<M_, blk, 0, stream>>>(
          ctmp + (size_t)c * M_ * D_, nullptr, 0.f,
          cln_w + (size_t)(l * C_ + c) * D_, cln_b + (size_t)(l * C_ + c) * D_,
          mixed, nullptr, c == 0 ? 0 : 1);
    }
    // x = LN(x + mixed/C)
    ln_fused<<<M_, blk, 0, stream>>>(x, mixed, 1.f / C_,
        ln1_w + (size_t)l * D_, ln1_b + (size_t)l * D_, x, xb, 0);

    // FFN: gate / up GEMMs
    gemm_bf16_wmma<<<dim3(FF_ / 64, M_ / 256, 1), blk, 0, stream>>>(
        xb, D_, 0, fg_b + (size_t)l * FF_ * D_, D_, 0,
        gbuf, FF_, 0, M_, FF_, D_, 0);
    gemm_bf16_wmma<<<dim3(FF_ / 64, M_ / 256, 1), blk, 0, stream>>>(
        xb, D_, 0, fu_b + (size_t)l * FF_ * D_, D_, 0,
        ubuf, FF_, 0, M_, FF_, D_, 0);
    // h = silu(gate) * up  (bf16)
    swiglu_kernel<<<2048, blk, 0, stream>>>(gbuf, ubuf, hb, (long)M_ * FF_);
    // y = h @ Wdown^T  (reuse ctmp as y)
    gemm_bf16_wmma<<<dim3(D_ / 64, M_ / 256, 1), blk, 0, stream>>>(
        hb, FF_, 0, fd_b + (size_t)l * D_ * FF_, FF_, 0,
        ctmp, D_, 0, M_, D_, FF_, 0);
    // x = LN(x + y)
    ln_fused<<<M_, blk, 0, stream>>>(x, ctmp, 1.f,
        ln2_w + (size_t)l * D_, ln2_b + (size_t)l * D_, x, xb, 0);
  }

  // final LN -> xb
  ln_fused<<<M_, blk, 0, stream>>>(x, nullptr, 0.f, fin_w, fin_b, x, xb, 0);

  // logits = x @ token_emb^T   (N = 50257, edge-guarded)
  gemm_bf16_wmma<<<dim3((V_ + 63) / 64, M_ / 256, 1), blk, 0, stream>>>(
      xb, D_, 0, temb_b, D_, 0,
      logits, V_, 0, M_, V_, D_, 0);
}
